// Attention_36137854828870
// MI455X (gfx1250) — matrix operand order repaired and checked by reading, not compiled, not run on hardware
//
#include <hip/hip_runtime.h>
#include <hip/hip_bf16.h>

typedef __attribute__((ext_vector_type(16))) __bf16 v16bf;
typedef __attribute__((ext_vector_type(8)))  float  v8f;
typedef __attribute__((ext_vector_type(4)))  unsigned int v4u;
typedef __attribute__((ext_vector_type(8)))  int  v8i;
typedef __attribute__((ext_vector_type(4)))  int  v4i;

namespace {
constexpr int B_  = 4;
constexpr int S_  = 2048;
constexpr int D_  = 1024;
constexpr int H_  = 16;
constexpr int HD_ = 64;
constexpr float SCL_ = 0.125f;        // 1/sqrt(64)
constexpr float NEGINF_ = -1.0e9f;

constexpr int BM = 128, BN = 128, BK = 32;
constexpr int LROW = 48;              // 32 bf16 payload + 16 bf16 TDM pad -> 96B row stride
}

union FragBF { v16bf v; uint4 u[2]; };

// ---- LDS byte-offset of a shared-memory pointer (ptrtoint of addrspace(3)) ----
typedef __attribute__((address_space(3))) void lds_void;
__device__ __forceinline__ unsigned lds_addr_of(void* p) {
    return (unsigned)(unsigned long long)(lds_void*)p;
}

// ---- Tensor Data Mover: 2D tile (rows x cols, bf16) global -> LDS, optional LDS pad ----
// pad_interval: 2^(v+1) DWORDs between pads; pad_amount: (v+1) DWORDs inserted. v<0 disables.
__device__ __forceinline__ void tdm_load_2d(unsigned lds_byte_off, const void* gaddr,
                                            int tensor_d0, int tensor_d1,
                                            int tile_d0, int tile_d1,
                                            long stride0_elems,
                                            int pad_interval, int pad_amount) {
    unsigned long long ga = (unsigned long long)gaddr;
    v4u g0;
    g0[0] = 1u;                                        // count=1, no gather
    g0[1] = lds_byte_off;                              // lds_addr
    g0[2] = (unsigned)(ga & 0xffffffffu);              // global_addr[31:0]
    g0[3] = (unsigned)((ga >> 32) & 0x01ffffffu) | 0x80000000u; // addr[56:32] | type=2
    unsigned w0 = (1u << 16);                          // data_size = 1 (2-byte elements)
    if (pad_amount >= 0)
        w0 |= (1u << 20) | ((unsigned)pad_interval << 22) | ((unsigned)pad_amount << 25);
    v8i g1;
    g1[0] = (int)w0;                                   // mask=0 | data_size | pad ctl
    g1[1] = (int)(((unsigned)tensor_d0 & 0xffffu) << 16);                 // dim0[15:0]
    g1[2] = (int)(((unsigned)tensor_d0 >> 16) | (((unsigned)tensor_d1 & 0xffffu) << 16));
    g1[3] = (int)(((unsigned)tensor_d1 >> 16) | ((unsigned)tile_d0 << 16));
    g1[4] = (int)((unsigned)tile_d1);                  // tile_dim1 | tile_dim2=0
    unsigned long long s0 = (unsigned long long)stride0_elems;
    g1[5] = (int)(s0 & 0xffffffffu);                   // tensor_dim0_stride[31:0]
    g1[6] = (int)((s0 >> 32) & 0xffffu);               // stride0[47:32] | stride1=0
    g1[7] = 0;
    v4i z4 = {};
    v8i z8 = {};
    // amdgpu-toolchain (clang-23) 6-arg form: (g0, g1, g2, g3, g4, cpol)
    __builtin_amdgcn_tensor_load_to_lds(g0, g1, z4, z4, z8, 0);
}

// A-fragment (16x32 bf16, MxK): lane row M=lane&15; K=(lane>=16?8:0)+{0..7} (elems 0..7)
// and +16 (elems 8..15). Two 16B loads.
__device__ __forceinline__ v16bf load_a_frag(const __bf16* base, int ld, int row0, int k0, int lane) {
    const int r  = row0 + (lane & 15);
    const int kb = k0 + ((lane >> 4) << 3);
    const __bf16* p = base + (size_t)r * ld + kb;
    FragBF f;
    f.u[0] = *reinterpret_cast<const uint4*>(p);
    f.u[1] = *reinterpret_cast<const uint4*>(p + 16);
    return f.v;
}

// B-fragment (32x16 bf16, KxN), source N-major: lane col N=lane&15; K=(lane>=16?8:0)+{0..7} (elems 0..7) and +16 (elems 8..15).
__device__ __forceinline__ v16bf load_b_frag(const __bf16* base, int ld, int n0, int k0, int lane) {
    const int n  = n0 + (lane & 15);
    const int kb = k0 + ((lane >> 4) << 3);
    const __bf16* p = base + (size_t)n * ld + kb;
    FragBF f;
    f.u[0] = *reinterpret_cast<const uint4*>(p);
    f.u[1] = *reinterpret_cast<const uint4*>(p + 16);
    return f.v;
}

// Same fragment gathers out of padded LDS tiles (row stride LROW bf16, 16B aligned)
__device__ __forceinline__ v16bf lds_a_frag(const __bf16* base, int row0, int lane) {
    const int r  = row0 + (lane & 15);
    const int kb = (lane >> 4) << 3;
    const __bf16* p = base + r * LROW + kb;
    FragBF f;
    f.u[0] = *reinterpret_cast<const uint4*>(p);
    f.u[1] = *reinterpret_cast<const uint4*>(p + 16);
    return f.v;
}
__device__ __forceinline__ v16bf lds_b_frag(const __bf16* base, int n0, int lane) {
    const int n  = n0 + (lane & 15);
    const int kb = (lane >> 4) << 3;
    const __bf16* p = base + n * LROW + kb;
    FragBF f;
    f.u[0] = *reinterpret_cast<const uint4*>(p);
    f.u[1] = *reinterpret_cast<const uint4*>(p + 16);
    return f.v;
}

// ---------------- conversion / relayout kernels ----------------

__global__ void cvt_f32_bf16(const float* __restrict__ in, __bf16* __restrict__ out) {
    int tid = blockIdx.x * blockDim.x + threadIdx.x;
    out[tid] = (__bf16)in[tid];
}

// wt[n*D + k] = (bf16) w[k*D + n]  (transpose to N-major for B-fragment loads)
__global__ void w_transpose_cvt(const float* __restrict__ w, __bf16* __restrict__ wt) {
    int tid = blockIdx.x * blockDim.x + threadIdx.x;   // D*D
    int k = tid & (D_ - 1);
    int n = tid >> 10;
    wt[tid] = (__bf16)w[(size_t)k * D_ + n];
}

// in:  [B*S, D] bf16 (col = h*64+d);  out: [B*H, S, 64] bf16, rotary applied in fp32
__global__ void rope_relayout(const __bf16* __restrict__ in, const float* __restrict__ fc,
                              const float* __restrict__ fs, __bf16* __restrict__ out) {
    int tid = blockIdx.x * blockDim.x + threadIdx.x;   // B*S*H*32
    int i = tid & 31;
    int h = (tid >> 5) & (H_ - 1);
    int s = (tid >> 9) & (S_ - 1);
    int b = tid >> 20;
    const __bf16* src = in + (size_t)(b * S_ + s) * D_ + h * HD_ + 2 * i;
    float t0 = (float)src[0];
    float t1 = (float)src[1];
    float c  = fc[s * 32 + i];
    float sn = fs[s * 32 + i];
    __bf16* dst = out + ((size_t)(b * H_ + h) * S_ + s) * HD_ + 2 * i;
    dst[0] = (__bf16)(t0 * c - t1 * sn);
    dst[1] = (__bf16)(t0 * sn + t1 * c);
}

// in: [B*S, D] bf16 -> out: [B*H, 64, S] bf16 (V transposed for PV B-fragments)
__global__ void v_transpose(const __bf16* __restrict__ in, __bf16* __restrict__ out) {
    int tid = blockIdx.x * blockDim.x + threadIdx.x;   // B*H*64*S
    int s = tid & (S_ - 1);
    int d = (tid >> 11) & (HD_ - 1);
    int h = (tid >> 17) & (H_ - 1);
    int b = tid >> 21;
    out[tid] = in[(size_t)(b * S_ + s) * D_ + h * HD_ + d];
}

// ------- WMMA GEMM, TDM double-buffered LDS pipeline: C[M,N] = A[M,K] * Bt[N,K]^T -------
// 256 threads = 8 waves; block tile 128x128; wave tile 32(M) x 64(N) = 2x4 accumulators.

template <bool OUT_F32>
__global__ __launch_bounds__(256) void gemm_bf16_wmma(
    const __bf16* __restrict__ A, const __bf16* __restrict__ Bt,
    void* __restrict__ C, int M, int N, int K) {
    __shared__ __bf16 smA[2][BM][LROW];   // 2 x 12KB (TDM pads 32->48 bf16 rows)
    __shared__ __bf16 smB[2][BN][LROW];   // 2 x 12KB

    const int lane = threadIdx.x & 31;
    const int wave = threadIdx.x >> 5;
    const int wm = wave >> 1;             // 0..3 -> 32-row band
    const int wn = wave & 1;              // 0..1 -> 64-col band
    const int m0 = blockIdx.y * BM;
    const int n0 = blockIdx.x * BN;

    unsigned la[2] = { lds_addr_of(&smA[0][0][0]), lds_addr_of(&smA[1][0][0]) };
    unsigned lb[2] = { lds_addr_of(&smB[0][0][0]), lds_addr_of(&smB[1][0][0]) };

    if (wave == 0) {   // TDM ignores EXEC; issue once per block (wave 0 only)
        tdm_load_2d(la[0], A  + (size_t)m0 * K, K, M, BK, BM, K, /*pad*/3, 7);
        tdm_load_2d(lb[0], Bt + (size_t)n0 * K, K, N, BK, BN, K, /*pad*/3, 7);
    }

    v8f acc[2][4] = {};
    int buf = 0;
    for (int k = 0; k < K; k += BK) {
        if (wave == 0) {
            if (k + BK < K) {
                tdm_load_2d(la[buf ^ 1], A  + (size_t)m0 * K + (k + BK), K, M, BK, BM, K, 3, 7);
                tdm_load_2d(lb[buf ^ 1], Bt + (size_t)n0 * K + (k + BK), K, N, BK, BN, K, 3, 7);
                __builtin_amdgcn_s_wait_tensorcnt(2);  // retire current buffer's pair
            } else {
                __builtin_amdgcn_s_wait_tensorcnt(0);
            }
        }
        __syncthreads();                  // current buffer visible to all waves

        const __bf16* Ab = &smA[buf][0][0];
        const __bf16* Bb = &smB[buf][0][0];
        v16bf af0 = lds_a_frag(Ab, wm * 32,      lane);
        v16bf af1 = lds_a_frag(Ab, wm * 32 + 16, lane);
        v16bf bf0 = lds_b_frag(Bb, wn * 64,      lane);
        v16bf bf1 = lds_b_frag(Bb, wn * 64 + 16, lane);
        v16bf bf2 = lds_b_frag(Bb, wn * 64 + 32, lane);
        v16bf bf3 = lds_b_frag(Bb, wn * 64 + 48, lane);
        acc[0][0] = __builtin_amdgcn_wmma_f32_16x16x32_bf16(false, af0, false, bf0, (short)0, acc[0][0], false, false);
        acc[0][1] = __builtin_amdgcn_wmma_f32_16x16x32_bf16(false, af0, false, bf1, (short)0, acc[0][1], false, false);
        acc[0][2] = __builtin_amdgcn_wmma_f32_16x16x32_bf16(false, af0, false, bf2, (short)0, acc[0][2], false, false);
        acc[0][3] = __builtin_amdgcn_wmma_f32_16x16x32_bf16(false, af0, false, bf3, (short)0, acc[0][3], false, false);
        acc[1][0] = __builtin_amdgcn_wmma_f32_16x16x32_bf16(false, af1, false, bf0, (short)0, acc[1][0], false, false);
        acc[1][1] = __builtin_amdgcn_wmma_f32_16x16x32_bf16(false, af1, false, bf1, (short)0, acc[1][1], false, false);
        acc[1][2] = __builtin_amdgcn_wmma_f32_16x16x32_bf16(false, af1, false, bf2, (short)0, acc[1][2], false, false);
        acc[1][3] = __builtin_amdgcn_wmma_f32_16x16x32_bf16(false, af1, false, bf3, (short)0, acc[1][3], false, false);

        __syncthreads();                  // all reads done before TDM overwrites this buffer
        buf ^= 1;
    }

    const int half8 = (lane >> 4) << 3;
    const int nidx  = lane & 15;
#pragma unroll
    for (int i = 0; i < 2; ++i) {
#pragma unroll
        for (int j = 0; j < 4; ++j) {
            const int col = n0 + wn * 64 + j * 16 + nidx;
            const int rb  = m0 + wm * 32 + i * 16 + half8;
            if constexpr (OUT_F32) {
                float* Cf = reinterpret_cast<float*>(C);
#pragma unroll
                for (int r = 0; r < 8; ++r) Cf[(size_t)(rb + r) * N + col] = acc[i][j][r];
            } else {
                __bf16* Cb = reinterpret_cast<__bf16*>(C);
#pragma unroll
                for (int r = 0; r < 8; ++r) Cb[(size_t)(rb + r) * N + col] = (__bf16)acc[i][j][r];
            }
        }
    }
}

// ---------------- flash attention (causal, online softmax) ----------------
// Q,K: [B*H, S, 64] bf16;  Vt: [B*H, 64, S] bf16;  O: [B*S, D] bf16.

__global__ __launch_bounds__(256) void flash_attn(
    const __bf16* __restrict__ Q, const __bf16* __restrict__ Kt,
    const __bf16* __restrict__ Vt, __bf16* __restrict__ O) {
    const int lane = threadIdx.x & 31;
    const int wave = threadIdx.x >> 5;
    const int bh = blockIdx.y;
    const int b  = bh >> 4;
    const int h  = bh & 15;
    const int q0 = (blockIdx.x * 8 + wave) * 16;

    // per-wave P staging: row stride 48 bf16 = 96B keeps b128 reads 16B-aligned
    __shared__ __bf16 pst[8][16][48];

    const __bf16* Qb = Q  + (size_t)bh * S_ * HD_;
    const __bf16* Kb = Kt + (size_t)bh * S_ * HD_;
    const __bf16* Vb = Vt + (size_t)bh * HD_ * S_;

    const v16bf aq0 = load_a_frag(Qb, HD_, q0, 0,  lane);
    const v16bf aq1 = load_a_frag(Qb, HD_, q0, 32, lane);

    v8f acc0 = {}, acc1 = {}, acc2 = {}, acc3 = {};
    float rowmax[8], rowsum[8];
#pragma unroll
    for (int r = 0; r < 8; ++r) { rowmax[r] = NEGINF_; rowsum[r] = 0.f; }

    const int half8 = (lane >> 4) << 3;    // 0 for lanes 0..15, 8 for 16..31
    const int nidx  = lane & 15;

    for (int kc = 0; kc <= q0 + 15; kc += 32) {
        // ---- scores for two 16-key tiles (keys kc..kc+15 and kc+16..kc+31) ----
        v8f s0 = {}, s1 = {};
        {
            v16bf bk = load_b_frag(Kb, HD_, kc, 0, lane);
            s0 = __builtin_amdgcn_wmma_f32_16x16x32_bf16(false, aq0, false, bk, (short)0, s0, false, false);
            bk = load_b_frag(Kb, HD_, kc, 32, lane);
            s0 = __builtin_amdgcn_wmma_f32_16x16x32_bf16(false, aq1, false, bk, (short)0, s0, false, false);
            bk = load_b_frag(Kb, HD_, kc + 16, 0, lane);
            s1 = __builtin_amdgcn_wmma_f32_16x16x32_bf16(false, aq0, false, bk, (short)0, s1, false, false);
            bk = load_b_frag(Kb, HD_, kc + 16, 32, lane);
            s1 = __builtin_amdgcn_wmma_f32_16x16x32_bf16(false, aq1, false, bk, (short)0, s1, false, false);
        }
        const int key0 = kc + nidx;
        const int key1 = kc + 16 + nidx;
        // ---- online softmax per row (C-layout rows: half8 + r) ----
#pragma unroll
        for (int r = 0; r < 8; ++r) {
            const int qidx = q0 + half8 + r;
            float v0 = s0[r] * SCL_ + ((key0 > qidx) ? NEGINF_ : 0.f);
            float v1 = s1[r] * SCL_ + ((key1 > qidx) ? NEGINF_ : 0.f);
            float mx = fmaxf(v0, v1);
#pragma unroll
            for (int m = 8; m >= 1; m >>= 1) mx = fmaxf(mx, __shfl_xor(mx, m, 32));
            const float nm   = fmaxf(rowmax[r], mx);
            const float corr = __expf(rowmax[r] - nm);
            rowmax[r] = nm;
            const float p0 = __expf(v0 - nm);
            const float p1 = __expf(v1 - nm);
            float rs = p0 + p1;
#pragma unroll
            for (int m = 8; m >= 1; m >>= 1) rs += __shfl_xor(rs, m, 32);
            rowsum[r] = rowsum[r] * corr + rs;
            acc0[r] *= corr; acc1[r] *= corr; acc2[r] *= corr; acc3[r] *= corr;
            pst[wave][half8 + r][nidx]      = (__bf16)p0;
            pst[wave][half8 + r][16 + nidx] = (__bf16)p1;
        }
        // order intra-wave LDS store -> load (no workgroup barrier: loop is divergent per wave)
        asm volatile("s_wait_dscnt 0x0" ::: "memory");
        // re-swizzle P tile (C-layout in LDS) into A-fragment layout
        FragBF pf;
        {
            const __bf16* pp = &pst[wave][nidx][half8];
            pf.u[0] = *reinterpret_cast<const uint4*>(pp);
            pf.u[1] = *reinterpret_cast<const uint4*>(pp + 16);
        }
        asm volatile("" ::: "memory");
        // ---- PV: acc[dt] += P (16x32) * V^T-tiles (32x16) ----
        v16bf bv = load_b_frag(Vb, S_, 0, kc, lane);
        acc0 = __builtin_amdgcn_wmma_f32_16x16x32_bf16(false, pf.v, false, bv, (short)0, acc0, false, false);
        bv = load_b_frag(Vb, S_, 16, kc, lane);
        acc1 = __builtin_amdgcn_wmma_f32_16x16x32_bf16(false, pf.v, false, bv, (short)0, acc1, false, false);
        bv = load_b_frag(Vb, S_, 32, kc, lane);
        acc2 = __builtin_amdgcn_wmma_f32_16x16x32_bf16(false, pf.v, false, bv, (short)0, acc2, false, false);
        bv = load_b_frag(Vb, S_, 48, kc, lane);
        acc3 = __builtin_amdgcn_wmma_f32_16x16x32_bf16(false, pf.v, false, bv, (short)0, acc3, false, false);
    }

    // ---- normalize and store to [B*S, D] bf16 (col = h*64 + d) ----
    __bf16* Ob = O + (size_t)b * S_ * D_ + (size_t)h * HD_;
#pragma unroll
    for (int r = 0; r < 8; ++r) {
        const float inv = 1.f / rowsum[r];
        const size_t row = (size_t)(q0 + half8 + r) * D_;
        Ob[row +  0 + nidx] = (__bf16)(acc0[r] * inv);
        Ob[row + 16 + nidx] = (__bf16)(acc1[r] * inv);
        Ob[row + 32 + nidx] = (__bf16)(acc2[r] * inv);
        Ob[row + 48 + nidx] = (__bf16)(acc3[r] * inv);
    }
}

// ---------------- host orchestration ----------------

extern "C" void kernel_launch(void* const* d_in, const int* in_sizes, int n_in,
                              void* d_out, int out_size, void* d_ws, size_t ws_size,
                              hipStream_t stream) {
    (void)in_sizes; (void)n_in; (void)out_size; (void)ws_size;
    const float* x  = (const float*)d_in[0];
    const float* fc = (const float*)d_in[1];
    const float* fs = (const float*)d_in[2];
    // d_in[3] = causal mask; computed inline in flash_attn
    const float* wq = (const float*)d_in[4];
    const float* wk = (const float*)d_in[5];
    const float* wv = (const float*)d_in[6];
    const float* wo = (const float*)d_in[7];

    char* ws = (char*)d_ws;
    constexpr size_t SZ_XD = (size_t)B_ * S_ * D_ * 2;   // bf16 [B*S, D]
    constexpr size_t SZ_W  = (size_t)D_ * D_ * 2;        // bf16 [D, D]
    const size_t oXB = 0;
    const size_t oWQ = oXB + SZ_XD;
    const size_t oWK = oWQ + SZ_W;
    const size_t oWV = oWK + SZ_W;
    const size_t oWO = oWV + SZ_W;
    const size_t oQP = oWO + SZ_W;
    const size_t oKP = oQP + SZ_XD;
    const size_t oVP = oKP + SZ_XD;
    const size_t oQ  = oVP + SZ_XD;
    const size_t oK  = oQ  + SZ_XD;
    const size_t oVT = oK  + SZ_XD;
    const size_t oAT = oQP;   // qp dead after rope -> reuse for attention output
    auto bfp = [&](size_t off) { return (__bf16*)(ws + off); };

    const int M = B_ * S_;

    cvt_f32_bf16<<<M * D_ / 256, 256, 0, stream>>>(x, bfp(oXB));
    w_transpose_cvt<<<D_ * D_ / 256, 256, 0, stream>>>(wq, bfp(oWQ));
    w_transpose_cvt<<<D_ * D_ / 256, 256, 0, stream>>>(wk, bfp(oWK));
    w_transpose_cvt<<<D_ * D_ / 256, 256, 0, stream>>>(wv, bfp(oWV));
    w_transpose_cvt<<<D_ * D_ / 256, 256, 0, stream>>>(wo, bfp(oWO));

    dim3 gg(D_ / BN, M / BM);
    gemm_bf16_wmma<false><<<gg, 256, 0, stream>>>(bfp(oXB), bfp(oWQ), bfp(oQP), M, D_, D_);
    gemm_bf16_wmma<false><<<gg, 256, 0, stream>>>(bfp(oXB), bfp(oWK), bfp(oKP), M, D_, D_);
    gemm_bf16_wmma<false><<<gg, 256, 0, stream>>>(bfp(oXB), bfp(oWV), bfp(oVP), M, D_, D_);

    rope_relayout<<<(B_ * S_ * H_ * 32) / 256, 256, 0, stream>>>(bfp(oQP), fc, fs, bfp(oQ));
    rope_relayout<<<(B_ * S_ * H_ * 32) / 256, 256, 0, stream>>>(bfp(oKP), fc, fs, bfp(oK));
    v_transpose<<<(B_ * H_ * HD_ * S_) / 256, 256, 0, stream>>>(bfp(oVP), bfp(oVT));

    flash_attn<<<dim3(S_ / 128, B_ * H_), 256, 0, stream>>>(bfp(oQ), bfp(oK), bfp(oVT), bfp(oAT));

    gemm_bf16_wmma<true><<<gg, 256, 0, stream>>>(bfp(oAT), bfp(oWO), d_out, M, D_, D_);
}
